// GVRNN_18545668784461
// MI455X (gfx1250) — matrix operand order repaired and checked by reading, not compiled, not run on hardware
//
#include <hip/hip_runtime.h>
#include <hip/hip_bf16.h>
#include <math.h>
#include <stdint.h>

// ---------------- problem constants ----------------
#define TT     10
#define BB     512
#define NAL    23
#define NAG    22
#define ZDIM   92
#define NEDGE  506      // 23*22, receiver-major contiguous blocks of 22
#define RNNH   64
#define INHPAD 1600     // 184 gnn (pad 192) + 22*64 micro
#define DECPAD 192      // 184 -> 192
#define DOUT   88       // X_DIM * N_AGENTS
#define LKL_OFF ((size_t)2 * TT * BB * DOUT)

typedef __attribute__((ext_vector_type(16))) _Float16 v16h;
typedef __attribute__((ext_vector_type(8)))  float    v8f;

// ---------------- small math helpers ----------------
__device__ __forceinline__ float eluf(float x)      { return x > 0.f ? x : (expf(x) - 1.f); }
__device__ __forceinline__ float sigmf(float x)     { return 1.f / (1.f + expf(-x)); }
__device__ __forceinline__ float softplusf(float x) { return x > 20.f ? x : log1pf(expf(x)); }

__device__ __forceinline__ float wave_sum(float v) {
#pragma unroll
  for (int m = 16; m > 0; m >>= 1) v += __shfl_xor(v, m, 32);
  return v;
}

__device__ __forceinline__ v8f v8f_zero() {
  v8f c;
#pragma unroll
  for (int i = 0; i < 8; ++i) c[i] = 0.f;
  return c;
}

// ---------------- CDNA5 async global->LDS copy (16B per lane) ----------
// GV mode: LDS dest address in VGPR (low 32 bits of generic shared ptr),
// 64-bit global address in VGPR pair. Tracked by ASYNCcnt.
__device__ __forceinline__ void async_copy16(void* lds_dst, const void* gsrc) {
  unsigned l = (unsigned)(uintptr_t)lds_dst;
  asm volatile("global_load_async_to_lds_b128 %0, %1, off"
               :: "v"(l), "v"(gsrc) : "memory");
}
__device__ __forceinline__ void async_wait0() {
  asm volatile("s_wait_asynccnt 0" ::: "memory");
}

// ---------------- WMMA fragments (CDNA5 16x16x32 f16) ----------------
// A 16x32 f16: lanes 0-15 row m, K k0..k0+15; lanes 16-31 K k0+16..k0+31.
__device__ __forceinline__ v16h load_fragA(const _Float16* __restrict__ base, int stride, int k0) {
  const int lane = threadIdx.x & 31;
  const _Float16* p = base + (lane & 15) * stride + k0 + ((lane >> 4) << 4);
  v16h a;
#pragma unroll
  for (int j = 0; j < 8; ++j) a[j] = p[j];
#pragma unroll
  for (int j = 0; j < 8; ++j) a[8 + j] = p[8 + j];
  return a;
}
// B 32x16 f16, column n per lane: lanes 0-15 K k0..k0+15; lanes 16-31 K k0+16..k0+31.
__device__ __forceinline__ v16h load_fragB(const _Float16* __restrict__ base, int stride, int k0) {
  const int lane = threadIdx.x & 31;
  const _Float16* p = base + (lane & 15) * stride + k0 + ((lane >> 4) << 4);
  v16h b;
#pragma unroll
  for (int j = 0; j < 16; ++j) b[j] = p[j];
  return b;
}
__device__ __forceinline__ v8f wmma_acc(v16h a, v16h b, v8f c) {
  return __builtin_amdgcn_wmma_f32_16x16x32_f16(false, a, false, b, (short)0, c, false, false);
}

// ---------------- deterministic BN stat reductions ----------------
__device__ __forceinline__ void block_stat_reduce(const float h2[8], float* __restrict__ partials) {
  __shared__ float part[8][16];
  const int wave = threadIdx.x >> 5, lane = threadIdx.x & 31;
#pragma unroll
  for (int o = 0; o < 8; ++o) {
    float s = wave_sum(h2[o]);
    float q = wave_sum(h2[o] * h2[o]);
    if (lane == 0) { part[wave][o] = s; part[wave][8 + o] = q; }
  }
  __syncthreads();
  if (threadIdx.x < 16) {
    float acc = 0.f;
#pragma unroll
    for (int w = 0; w < 8; ++w) acc += part[w][threadIdx.x];
    partials[blockIdx.x * 16 + threadIdx.x] = acc;
  }
}

__device__ __forceinline__ void load_bn_stats(const float* __restrict__ partials, int nblocks,
                                              float cnt, const float* __restrict__ g,
                                              const float* __restrict__ be,
                                              float* __restrict__ sc, float* __restrict__ sh) {
  __shared__ float raw[16];
  if (threadIdx.x < 16) {
    float acc = 0.f;
    for (int i = 0; i < nblocks; ++i) acc += partials[i * 16 + threadIdx.x];
    raw[threadIdx.x] = acc;
  }
  __syncthreads();
  if (threadIdx.x < 8) {
    const int c = threadIdx.x;
    float m  = raw[c] / cnt;
    float v  = raw[8 + c] / cnt - m * m;
    float rs = rsqrtf(v + 1e-5f);
    float scale = rs * g[c];
    sc[c] = scale;
    sh[c] = be[c] - m * scale;
  }
  __syncthreads();
}

// ---------------- prep: fp16 weight packing + state init ----------------
__global__ void __launch_bounds__(256) prep_kernel(
    const float* __restrict__ pmW, const float* __restrict__ psW,
    const float* __restrict__ emW, const float* __restrict__ esW,
    const float* __restrict__ dmW, const float* __restrict__ dsW,
    const float* __restrict__ Wih, const float* __restrict__ Whh,
    _Float16* __restrict__ Wp,   _Float16* __restrict__ decWp,
    _Float16* __restrict__ Wihp, _Float16* __restrict__ Whhp,
    _Float16* __restrict__ priF, _Float16* __restrict__ encF,
    _Float16* __restrict__ dgF) {
  const int tid0 = blockIdx.x * 256 + threadIdx.x;
  const int strd = gridDim.x * 256;
  for (int i = tid0; i < 4 * 96 * INHPAD; i += strd) {
    int mat = i / (96 * INHPAD), rem = i % (96 * INHPAD);
    int n = rem / INHPAD, k = rem % INHPAD;
    const float* W = (mat == 0) ? pmW : (mat == 1) ? psW : (mat == 2) ? emW : esW;
    float v = 0.f;
    if (n < ZDIM) { if (k < 184) v = W[n * 1592 + k]; else if (k >= 192) v = W[n * 1592 + (k - 8)]; }
    Wp[i] = (_Float16)v;
  }
  for (int i = tid0; i < 2 * 96 * DECPAD; i += strd) {
    int mat = i / (96 * DECPAD), rem = i % (96 * DECPAD);
    int n = rem / DECPAD, k = rem % DECPAD;
    const float* W = mat ? dsW : dmW;
    decWp[i] = (_Float16)((n < DOUT && k < 184) ? W[n * 184 + k] : 0.f);
  }
  for (int i = tid0; i < NAG * 192 * 32; i += strd) {
    int a = i / (192 * 32), rem = i % (192 * 32);
    int n = rem / 32, k = rem % 32;
    Wihp[i] = (_Float16)((k < 8) ? Wih[(a * 192 + n) * 8 + k] : 0.f);
  }
  for (int i = tid0; i < NAG * 192 * 64; i += strd) Whhp[i] = (_Float16)Whh[i];
  for (int i = tid0; i < BB * INHPAD; i += strd) { priF[i] = (_Float16)0.f; encF[i] = (_Float16)0.f; }
  for (int i = tid0; i < BB * DECPAD; i += strd) dgF[i] = (_Float16)0.f;
}

// ---------------- GNN stage 1: mlp1 (4->8->8, pre-BN) ----------------
__global__ void __launch_bounds__(256) gnn_mlp1(
    const float* __restrict__ xin,
    const float* __restrict__ W1, const float* __restrict__ b1,
    const float* __restrict__ W2, const float* __restrict__ b2,
    float* __restrict__ hout, float* __restrict__ partials) {
  const int row = blockIdx.x * 256 + threadIdx.x;  // exact: 46*256 == 512*23
  const int b = row / NAL, n = row % NAL;
  float x[4], h1[8], h2[8];
#pragma unroll
  for (int c = 0; c < 4; ++c) x[c] = xin[b * ZDIM + n * 4 + c];
#pragma unroll
  for (int o = 0; o < 8; ++o) {
    float s = b1[o];
#pragma unroll
    for (int c = 0; c < 4; ++c) s += W1[o * 4 + c] * x[c];
    h1[o] = eluf(s);
  }
#pragma unroll
  for (int o = 0; o < 8; ++o) {
    float s = b2[o];
#pragma unroll
    for (int c = 0; c < 8; ++c) s += W2[o * 8 + c] * h1[c];
    h2[o] = eluf(s);
    hout[(size_t)row * 8 + o] = h2[o];
  }
  block_stat_reduce(h2, partials);
}

// ---------------- GNN stage 2: node2edge + mlp2 (18->8->8, pre-BN) ------
__global__ void __launch_bounds__(256) gnn_mlp2(
    const float* __restrict__ n1, const float* __restrict__ part1,
    const float* __restrict__ g1, const float* __restrict__ be1,
    const float* __restrict__ W1, const float* __restrict__ b1,
    const float* __restrict__ W2, const float* __restrict__ b2,
    float* __restrict__ ebuf, float* __restrict__ part2) {
  __shared__ float sc1[8], sh1[8];
  load_bn_stats(part1, 46, (float)(BB * NAL), g1, be1, sc1, sh1);
  const int row = blockIdx.x * 256 + threadIdx.x;  // exact: 1012*256 == 512*506
  const int b = row / NEDGE, e = row % NEDGE;
  const int r = e / NAG, i = e % NAG;
  const int s = (i < r) ? i : i + 1;
  float in[18];
  const float* psn = n1 + (size_t)(b * NAL + s) * 8;
  const float* prn = n1 + (size_t)(b * NAL + r) * 8;
#pragma unroll
  for (int c = 0; c < 8; ++c) in[c]     = psn[c] * sc1[c] + sh1[c];
#pragma unroll
  for (int c = 0; c < 8; ++c) in[8 + c] = prn[c] * sc1[c] + sh1[c];
  in[16] = (i == 0 && r >= 1 && r <= 21) ? 1.f : 0.f;  // EDGE_TYPE col 0
  in[17] = (r == NAG) ? 1.f : 0.f;                     // EDGE_TYPE col 1
  float h1[8], h2[8];
#pragma unroll
  for (int o = 0; o < 8; ++o) {
    float sacc = b1[o];
#pragma unroll
    for (int c = 0; c < 18; ++c) sacc += W1[o * 18 + c] * in[c];
    h1[o] = eluf(sacc);
  }
#pragma unroll
  for (int o = 0; o < 8; ++o) {
    float sacc = b2[o];
#pragma unroll
    for (int c = 0; c < 8; ++c) sacc += W2[o * 8 + c] * h1[c];
    h2[o] = eluf(sacc);
    ebuf[(size_t)row * 8 + o] = h2[o];
  }
  block_stat_reduce(h2, part2);
}

// ---------------- GNN stage 3: edge2node gather + mlp3 (10->8->8) -------
__global__ void __launch_bounds__(256) gnn_mlp3(
    const float* __restrict__ ebuf, const float* __restrict__ part2,
    const float* __restrict__ g2, const float* __restrict__ be2,
    const float* __restrict__ W1, const float* __restrict__ b1,
    const float* __restrict__ W2, const float* __restrict__ b2,
    float* __restrict__ n3buf, float* __restrict__ part3) {
  __shared__ float sc2[8], sh2[8];
  load_bn_stats(part2, 1012, (float)(BB * NEDGE), g2, be2, sc2, sh2);
  const int row = blockIdx.x * 256 + threadIdx.x;  // exact 512*23
  const int b = row / NAL, n = row % NAL;
  float inc[8];
#pragma unroll
  for (int c = 0; c < 8; ++c) inc[c] = 0.f;
  const float* pe = ebuf + ((size_t)b * NEDGE + n * NAG) * 8;
  for (int i = 0; i < NAG; ++i)
#pragma unroll
    for (int c = 0; c < 8; ++c) inc[c] += pe[i * 8 + c];
  float in[10];
#pragma unroll
  for (int c = 0; c < 8; ++c) in[c] = (inc[c] * sc2[c] + (float)NAG * sh2[c]) * (1.f / (float)NAL);
  in[8] = (n < NAG) ? (1.f / (float)NAL) : 0.f;
  in[9] = (n == NAG) ? (1.f / (float)NAL) : 0.f;
  float h1[8], h2[8];
#pragma unroll
  for (int o = 0; o < 8; ++o) {
    float s = b1[o];
#pragma unroll
    for (int c = 0; c < 10; ++c) s += W1[o * 10 + c] * in[c];
    h1[o] = eluf(s);
  }
#pragma unroll
  for (int o = 0; o < 8; ++o) {
    float s = b2[o];
#pragma unroll
    for (int c = 0; c < 8; ++c) s += W2[o * 8 + c] * h1[c];
    h2[o] = eluf(s);
    n3buf[(size_t)row * 8 + o] = h2[o];
  }
  block_stat_reduce(h2, part3);
}

// ---------------- GNN stage 4: BN3 apply, pack fp16 into GEMM operand ---
__global__ void __launch_bounds__(256) gnn_bn3_pack(
    const float* __restrict__ n3buf, const float* __restrict__ part3,
    const float* __restrict__ g3, const float* __restrict__ be3,
    _Float16* __restrict__ fbuf, int fstride) {
  __shared__ float sc[8], sh[8];
  load_bn_stats(part3, 46, (float)(BB * NAL), g3, be3, sc, sh);
  const int row = blockIdx.x * 256 + threadIdx.x;
  const int b = row / NAL, n = row % NAL;
#pragma unroll
  for (int o = 0; o < 8; ++o) {
    float v = n3buf[(size_t)row * 8 + o] * sc[o] + sh[o];
    fbuf[(size_t)b * fstride + n * 8 + o] = (_Float16)v;
  }
}

// ---------------- big linears: 4x [16,1600]@[1600,96] via WMMA ----------
// A-tile staged once per phase into LDS via async global->LDS b128 copies;
// weights stream from global (L2-resident) with prefetch.
__global__ void __launch_bounds__(256) big_linear_wmma(
    const _Float16* __restrict__ priF, const _Float16* __restrict__ encF,
    const _Float16* __restrict__ Wp,
    const float* __restrict__ pmb, const float* __restrict__ psb,
    const float* __restrict__ emb, const float* __restrict__ esb,
    float* __restrict__ pm, float* __restrict__ ps,
    float* __restrict__ em, float* __restrict__ es) {
  __shared__ _Float16 ftile[16][INHPAD];          // 50 KB activation tile
  const int b0 = blockIdx.x * 16;
  const int wave = threadIdx.x >> 5, lane = threadIdx.x & 31;
  for (int ph = 0; ph < 2; ++ph) {                // ph0: prior mats, ph1: encoder mats
    const _Float16* feat = (ph ? encF : priF) + (size_t)b0 * INHPAD;
    // ---- async stage: 16 rows x 1600 halves = 3200 x 16B chunks ----
    for (int i = threadIdx.x; i < 16 * (INHPAD / 8); i += 256) {
      const int m = i / (INHPAD / 8), c = (i % (INHPAD / 8)) * 8;
      async_copy16(&ftile[m][c], feat + (size_t)m * INHPAD + c);
    }
    async_wait0();
    __syncthreads();
    // ---- 12 N-tile jobs (2 matrices x 6 tiles) over 8 waves ----
    for (int job = wave; job < 12; job += 8) {
      const int mat = ph * 2 + job / 6, n0 = (job % 6) * 16;
      const _Float16* w = Wp + (size_t)mat * 96 * INHPAD + (size_t)n0 * INHPAD;
      v8f c = v8f_zero();
      for (int k0 = 0; k0 < INHPAD; k0 += 32) {
        __builtin_prefetch(w + (lane & 15) * INHPAD + k0 + 128, 0, 1);
        c = wmma_acc(load_fragA(&ftile[0][0], INHPAD, k0), load_fragB(w, INHPAD, k0), c);
      }
      const int n = n0 + (lane & 15);
      if (n < ZDIM) {
        const float* bias = (mat == 0) ? pmb : (mat == 1) ? psb : (mat == 2) ? emb : esb;
        float* dst        = (mat == 0) ? pm  : (mat == 1) ? ps  : (mat == 2) ? em  : es;
        const bool isstd = (mat & 1);
        const float bv = bias[n];
        const int mb = (lane >> 4) * 8;
#pragma unroll
        for (int r = 0; r < 8; ++r) {
          float v = c[r] + bv;
          if (isstd) v = softplusf(v);
          dst[(size_t)(b0 + mb + r) * ZDIM + n] = v;
        }
      }
    }
    __syncthreads();                              // protect ftile before restage
  }
}

// ---------------- z sample + per-block KLD partial ----------------------
__global__ void __launch_bounds__(256) z_kld_kernel(
    const float* __restrict__ em, const float* __restrict__ es,
    const float* __restrict__ pm, const float* __restrict__ ps,
    const float* __restrict__ eps_t, float* __restrict__ zbuf,
    float* __restrict__ klpart) {
  const int idx = blockIdx.x * 256 + threadIdx.x;  // exact 184*256 == 512*92
  const float em_ = em[idx], es_ = es[idx], pm_ = pm[idx], ps_ = ps[idx];
  zbuf[idx] = em_ + es_ * eps_t[idx];
  const float d = em_ - pm_;
  float kl = 0.5f * (2.f * logf(ps_ / es_) + (es_ * es_ + d * d) / (ps_ * ps_) - 1.f);
  float s = wave_sum(kl);
  __shared__ float wsum[8];
  if ((threadIdx.x & 31) == 0) wsum[threadIdx.x >> 5] = s;
  __syncthreads();
  if (threadIdx.x == 0) {
    float acc = 0.f;
#pragma unroll
    for (int w = 0; w < 8; ++w) acc += wsum[w];
    klpart[blockIdx.x] = acc;
  }
}

__global__ void __launch_bounds__(256) lkl_final(const float* __restrict__ klpart,
                                                 float* __restrict__ out_lkl) {
  float acc = 0.f;
  for (int i = threadIdx.x; i < TT * 184; i += 256) acc += klpart[i];
  acc = wave_sum(acc);
  __shared__ float wsum[8];
  if ((threadIdx.x & 31) == 0) wsum[threadIdx.x >> 5] = acc;
  __syncthreads();
  if (threadIdx.x == 0) {
    float t = 0.f;
#pragma unroll
    for (int w = 0; w < 8; ++w) t += wsum[w];
    *out_lkl = t;
  }
}

// ---------------- decoder linears: [16,192]@[192,96]x2 via WMMA ---------
__global__ void __launch_bounds__(256) dec_wmma(
    const _Float16* __restrict__ dgF, const _Float16* __restrict__ decWp,
    const float* __restrict__ dmb, const float* __restrict__ dsb,
    float* __restrict__ out, int t) {
  __shared__ _Float16 ftile[16][DECPAD];          // 6 KB
  const int b0 = blockIdx.x * 16;
  const int wave = threadIdx.x >> 5, lane = threadIdx.x & 31;
  {
    const _Float16* feat = dgF + (size_t)b0 * DECPAD;
    for (int i = threadIdx.x; i < 16 * (DECPAD / 8); i += 256) {
      const int m = i / (DECPAD / 8), c = (i % (DECPAD / 8)) * 8;
      async_copy16(&ftile[m][c], feat + (size_t)m * DECPAD + c);
    }
    async_wait0();
    __syncthreads();
  }
  for (int job = wave; job < 12; job += 8) {
    const int mat = job / 6, n0 = (job % 6) * 16;
    const _Float16* w = decWp + (size_t)mat * 96 * DECPAD + (size_t)n0 * DECPAD;
    v8f c = v8f_zero();
    for (int k0 = 0; k0 < DECPAD; k0 += 32)
      c = wmma_acc(load_fragA(&ftile[0][0], DECPAD, k0), load_fragB(w, DECPAD, k0), c);
    const int n = n0 + (lane & 15);
    if (n < DOUT) {
      const float bv = (mat ? dsb : dmb)[n];
      const size_t basei = (mat ? (size_t)TT * BB * DOUT : 0) + (size_t)t * BB * DOUT;
      const int mb = (lane >> 4) * 8;
#pragma unroll
      for (int r = 0; r < 8; ++r) {
        float v = c[r] + bv;
        if (mat) v = softplusf(v);
        out[basei + (size_t)(b0 + mb + r) * DOUT + n] = v;
      }
    }
  }
}

// ---------------- per-agent GRU via WMMA (gi and gh kept separate) ------
__global__ void __launch_bounds__(256) gru_wmma(
    const float* __restrict__ xt, const float* __restrict__ zbuf,
    _Float16* __restrict__ priF, _Float16* __restrict__ encF,
    const _Float16* __restrict__ Wihp, const _Float16* __restrict__ Whhp,
    const float* __restrict__ bih, const float* __restrict__ bhh) {
  __shared__ _Float16 Agin[16][32];   // [x(4), z(4), 0-pad] fp16
  __shared__ _Float16 Ah[16][64];     // h_t fp16
  __shared__ float gi[16][192];
  __shared__ float gh[16][192];
  const int a  = blockIdx.x / 32;
  const int b0 = (blockIdx.x % 32) * 16;
  const int tid = threadIdx.x;
  for (int i = tid; i < 16 * 32; i += 256) {
    const int m = i >> 5, cc = i & 31;
    const int b = b0 + m;
    float v = 0.f;
    if (cc < 4)      v = xt[b * ZDIM + a * 4 + cc];
    else if (cc < 8) v = zbuf[b * ZDIM + a * 4 + (cc - 4)];
    Agin[m][cc] = (_Float16)v;
  }
  {
    // async stage h tile: 16 rows x 64 halves = 128 x 16B chunks
    const _Float16* gsrc = priF + (size_t)b0 * INHPAD + 192 + (size_t)a * 64;
    for (int i = tid; i < 16 * 8; i += 256) {
      const int m = i >> 3, c = (i & 7) * 8;
      async_copy16(&Ah[m][c], gsrc + (size_t)m * INHPAD + c);
    }
    async_wait0();
  }
  __syncthreads();
  const int wave = tid >> 5, lane = tid & 31;
  for (int nt = wave; nt < 12; nt += 8) {       // 192 gate cols = 12 N-tiles
    const int n0 = nt * 16;
    v8f ci = v8f_zero(), ch = v8f_zero();
    ci = wmma_acc(load_fragA(&Agin[0][0], 32, 0),
                  load_fragB(Wihp + ((size_t)a * 192 + n0) * 32, 32, 0), ci);
    for (int k0 = 0; k0 < 64; k0 += 32)
      ch = wmma_acc(load_fragA(&Ah[0][0], 64, k0),
                    load_fragB(Whhp + ((size_t)a * 192 + n0) * 64, 64, k0), ch);
    const int n = n0 + (lane & 15);
    const int mb = (lane >> 4) * 8;
    const float bi = bih[a * 192 + n], bh = bhh[a * 192 + n];
#pragma unroll
    for (int r = 0; r < 8; ++r) { gi[mb + r][n] = ci[r] + bi; gh[mb + r][n] = ch[r] + bh; }
  }
  __syncthreads();
  for (int i = tid; i < 16 * 64; i += 256) {
    const int m = i >> 6, j = i & 63;
    const float rr = sigmf(gi[m][j] + gh[m][j]);
    const float zz = sigmf(gi[m][64 + j] + gh[m][64 + j]);
    const float nn = tanhf(gi[m][128 + j] + rr * gh[m][128 + j]);
    const float ho = (float)Ah[m][j];
    const _Float16 hn = (_Float16)((1.f - zz) * nn + zz * ho);
    const size_t idx = (size_t)(b0 + m) * INHPAD + 192 + a * 64 + j;
    priF[idx] = hn; encF[idx] = hn;   // h feeds both GEMM operands next step
  }
}

// ---------------- host orchestration ----------------
extern "C" void kernel_launch(void* const* d_in, const int* in_sizes, int n_in,
                              void* d_out, int out_size, void* d_ws, size_t ws_size,
                              hipStream_t stream) {
  const float* states = (const float*)d_in[0];
  const float* eps    = (const float*)d_in[1];
  const float* m1W1 = (const float*)d_in[2],  *m1b1 = (const float*)d_in[3];
  const float* m1W2 = (const float*)d_in[4],  *m1b2 = (const float*)d_in[5];
  const float* m1g  = (const float*)d_in[6],  *m1be = (const float*)d_in[7];
  const float* m2W1 = (const float*)d_in[8],  *m2b1 = (const float*)d_in[9];
  const float* m2W2 = (const float*)d_in[10], *m2b2 = (const float*)d_in[11];
  const float* m2g  = (const float*)d_in[12], *m2be = (const float*)d_in[13];
  const float* m3W1 = (const float*)d_in[14], *m3b1 = (const float*)d_in[15];
  const float* m3W2 = (const float*)d_in[16], *m3b2 = (const float*)d_in[17];
  const float* m3g  = (const float*)d_in[18], *m3be = (const float*)d_in[19];
  const float* encmW = (const float*)d_in[20], *encmb = (const float*)d_in[21];
  const float* encsW = (const float*)d_in[22], *encsb = (const float*)d_in[23];
  const float* primW = (const float*)d_in[24], *primb = (const float*)d_in[25];
  const float* prisW = (const float*)d_in[26], *prisb = (const float*)d_in[27];
  const float* decmW = (const float*)d_in[28], *decmb = (const float*)d_in[29];
  const float* decsW = (const float*)d_in[30], *decsb = (const float*)d_in[31];
  const float* gWih  = (const float*)d_in[32], *gWhh  = (const float*)d_in[33];
  const float* gbih  = (const float*)d_in[34], *gbhh  = (const float*)d_in[35];

  char* base = (char*)d_ws;
  size_t off = 0;
  auto take = [&](size_t bytes) -> void* {
    void* p = base + off;
    off = (off + bytes + 255) & ~(size_t)255;
    return p;
  };
  _Float16* Wp    = (_Float16*)take((size_t)4 * 96 * INHPAD * 2);
  _Float16* decWp = (_Float16*)take((size_t)2 * 96 * DECPAD * 2);
  _Float16* Wihp  = (_Float16*)take((size_t)NAG * 192 * 32 * 2);
  _Float16* Whhp  = (_Float16*)take((size_t)NAG * 192 * 64 * 2);
  _Float16* priF  = (_Float16*)take((size_t)BB * INHPAD * 2);
  _Float16* encF  = (_Float16*)take((size_t)BB * INHPAD * 2);
  _Float16* dgF   = (_Float16*)take((size_t)BB * DECPAD * 2);
  float* n1buf  = (float*)take((size_t)BB * NAL * 8 * 4);
  float* ebuf   = (float*)take((size_t)BB * NEDGE * 8 * 4);
  float* n3buf  = (float*)take((size_t)BB * NAL * 8 * 4);
  float* PA     = (float*)take((size_t)46 * 16 * 4);
  float* PB     = (float*)take((size_t)1012 * 16 * 4);
  float* pm     = (float*)take((size_t)BB * ZDIM * 4);
  float* ps     = (float*)take((size_t)BB * ZDIM * 4);
  float* em     = (float*)take((size_t)BB * ZDIM * 4);
  float* es     = (float*)take((size_t)BB * ZDIM * 4);
  float* zbuf   = (float*)take((size_t)BB * ZDIM * 4);
  float* klpart = (float*)take((size_t)TT * 184 * 4);

  float* out = (float*)d_out;

  prep_kernel<<<512, 256, 0, stream>>>(primW, prisW, encmW, encsW, decmW, decsW, gWih, gWhh,
                                       Wp, decWp, Wihp, Whhp, priF, encF, dgF);

  auto run_gnn = [&](const float* xin, int p, _Float16* fbuf, int fstride) {
    gnn_mlp1<<<46, 256, 0, stream>>>(xin, m1W1 + p * 8 * 4, m1b1 + p * 8,
                                     m1W2 + p * 64, m1b2 + p * 8, n1buf, PA);
    gnn_mlp2<<<1012, 256, 0, stream>>>(n1buf, PA, m1g + p * 8, m1be + p * 8,
                                       m2W1 + p * 8 * 18, m2b1 + p * 8,
                                       m2W2 + p * 64, m2b2 + p * 8, ebuf, PB);
    gnn_mlp3<<<46, 256, 0, stream>>>(ebuf, PB, m2g + p * 8, m2be + p * 8,
                                     m3W1 + p * 8 * 10, m3b1 + p * 8,
                                     m3W2 + p * 64, m3b2 + p * 8, n3buf, PA);
    gnn_bn3_pack<<<46, 256, 0, stream>>>(n3buf, PA, m3g + p * 8, m3be + p * 8, fbuf, fstride);
  };

  for (int t = 0; t < TT; ++t) {
    const float* y  = states + (size_t)t * BB * ZDIM;
    const float* x1 = states + (size_t)(t + 1) * BB * ZDIM;
    run_gnn(y,  0, priF, INHPAD);                       // prior GNN -> priF[:, :184]
    run_gnn(x1, 1, encF, INHPAD);                       // encoder GNN -> encF[:, :184]
    big_linear_wmma<<<32, 256, 0, stream>>>(priF, encF, Wp, primb, prisb, encmb, encsb,
                                            pm, ps, em, es);
    z_kld_kernel<<<184, 256, 0, stream>>>(em, es, pm, ps, eps + (size_t)t * BB * ZDIM,
                                          zbuf, klpart + t * 184);
    run_gnn(zbuf, 2, dgF, DECPAD);                      // decoder GNN -> dgF
    dec_wmma<<<32, 256, 0, stream>>>(dgF, decWp, decmb, decsb, out, t);
    gru_wmma<<<704, 256, 0, stream>>>(x1, zbuf, priF, encF, Wihp, Whhp, gbih, gbhh);
  }
  lkl_final<<<1, 256, 0, stream>>>(klpart, out + LKL_OFF);
}
